// DeformConv2d_40114994544721
// MI455X (gfx1250) — matrix operand order repaired and checked by reading, not compiled, not run on hardware
//
#include <hip/hip_runtime.h>
#include <stdint.h>

// ---------------- problem constants ----------------
#define Bsz 8
#define Cch 128
#define Hh  64
#define Ww  64
#define Och 128
#define KT  9
#define HWSZ  (Hh * Ww)            // 4096
#define HWC   (Hh * Ww * Cch)      // 524288
#define SLAB  (Och * Cch)          // 16384 bf16 = 32 KB per tap

typedef __attribute__((ext_vector_type(16))) __bf16 v16bf;
typedef __attribute__((ext_vector_type(8)))  __bf16 v8bf;
typedef __attribute__((ext_vector_type(2)))  __bf16 v2bf;
typedef __attribute__((ext_vector_type(8)))  float  v8f;
typedef __attribute__((ext_vector_type(4)))  unsigned int u32x4;
typedef __attribute__((ext_vector_type(8)))  int i32x8;
typedef __attribute__((ext_vector_type(4)))  int i32x4;

#if __has_builtin(__builtin_amdgcn_tensor_load_to_lds)
#define HAVE_TDM 1
#else
#define HAVE_TDM 0
#endif
#if __has_builtin(__builtin_amdgcn_s_wait_tensorcnt)
#define WAIT_TENSOR(n) __builtin_amdgcn_s_wait_tensorcnt((short)(n))
#else
#define WAIT_TENSOR(n)
#endif

// bf16 <-> f32 helpers via bit manipulation (RNE on pack)
__device__ __forceinline__ float bfhi2f(unsigned int u) {           // high 16 bits
    return __builtin_bit_cast(float, u & 0xFFFF0000u);
}
__device__ __forceinline__ float bflo2f(unsigned int u) {           // low 16 bits
    return __builtin_bit_cast(float, u << 16);
}
__device__ __forceinline__ unsigned short f2bf(float f) {
    unsigned int u = __builtin_bit_cast(unsigned int, f);
    u += 0x7FFFu + ((u >> 16) & 1u);   // round-to-nearest-even
    return (unsigned short)(u >> 16);
}
// pack two f32 -> packed bf16 dword (single v_cvt_pk_bf16_f32 when available)
__device__ __forceinline__ unsigned int pk_bf16(float a, float b) {
#if __has_builtin(__builtin_amdgcn_cvt_pk_bf16_f32)
    v2bf p = __builtin_amdgcn_cvt_pk_bf16_f32(a, b);
    return __builtin_bit_cast(unsigned int, p);
#else
    return (unsigned int)f2bf(a) | ((unsigned int)f2bf(b) << 16);
#endif
}

#if HAVE_TDM
// Issue one TDM load of a contiguous 32 KB slab (16384 x bf16) into LDS.
__device__ __forceinline__ void tdm_load_slab(unsigned lds_addr, const unsigned short* gsrc) {
    unsigned long long ga = (unsigned long long)(uintptr_t)gsrc;
    u32x4 g0 = { 1u,                                   // count=1 (valid), user mode
                 lds_addr,                             // D#.lds_addr
                 (unsigned)(ga & 0xFFFFFFFFu),         // global_addr[31:0]
                 ((unsigned)(ga >> 32) & 0x01FFFFFFu)  // global_addr[56:32]
                 | 0x80000000u };                      // type=2 ("image")
    i32x8 g1 = { (int)0x00010000,    // wg_mask=0, data_size=1 (2B)
                 (int)0x40000000,    // tensor_dim0 = 16384 (low16 in [63:48])
                 (int)0x00010000,    // tensor_dim1 = 1 (low16 in [95:80])
                 (int)0x40000000,    // tile_dim0 = 16384 in [127:112]
                 (int)0x00000001,    // tile_dim1 = 1, tile_dim2 = 0
                 (int)16384,         // tensor_dim0_stride low32
                 0, 0 };             // remaining strides = 0
    i32x4 z4 = {0, 0, 0, 0};
#if __clang_major__ >= 23
    i32x8 z8 = {0, 0, 0, 0, 0, 0, 0, 0};
    __builtin_amdgcn_tensor_load_to_lds(g0, g1, z4, z4, z8, 0);
#else
    __builtin_amdgcn_tensor_load_to_lds(g0, g1, z4, z4, 0);
#endif
}
#endif

// ---------------- prep kernel 1: x [B,C,H,W] f32 -> xt [B,H,W,C] bf16 ----------------
__global__ void __launch_bounds__(256)
xpose_kernel(const float* __restrict__ x, unsigned short* __restrict__ xt) {
    int i = blockIdx.x * 256 + threadIdx.x;        // i = ((b*H + h)*W + w)*C + c
    int c = i & (Cch - 1);
    int t = i >> 7;
    int w = t & (Ww - 1); t >>= 6;
    int h = t & (Hh - 1);
    int b = t >> 6;
    float v = x[(((size_t)b * Cch + c) * Hh + h) * Ww + w];
    xt[i] = f2bf(v);
}

// ---------------- prep kernel 2: weight [O,C,K] f32 -> Wb [k][o][c] bf16 ----------------
__global__ void __launch_bounds__(256)
wprep_kernel(const float* __restrict__ wgt, unsigned short* __restrict__ wb) {
    int i = blockIdx.x * 256 + threadIdx.x;        // i = (k*O + o)*C + c
    if (i >= KT * Och * Cch) return;
    int c = i & (Cch - 1);
    int t = i >> 7;
    int o = t & (Och - 1);
    int k = t >> 7;
    wb[i] = f2bf(wgt[((size_t)o * Cch + c) * KT + k]);
}

// ---------------- fused deformable-sample + bf16 WMMA GEMM ----------------
// Block: 128 threads (4 waves). Owns 16 consecutive pixels (same b, same ho row)
// x all 128 output channels. GEMM roles: A = weight tile (16 o x 32 c),
// B = sampled tile (32 c x 16 pixels) -> D rows = o, cols = pixel, giving
// coalesced epilogue stores. Weight slabs arrive via TDM, double-buffered.
__global__ void __launch_bounds__(128)
dcn_kernel(const unsigned short* __restrict__ xt,
           const float*          __restrict__ offs,
           const unsigned short* __restrict__ wb,
           const float*          __restrict__ bias,
           float*                __restrict__ out) {
    __shared__ __align__(32) unsigned short Bl[2][SLAB];   // 64 KB double-buffered weights [o][c]
    __shared__ __align__(32) unsigned short Al[16 * Cch];  // 4 KB sampled tile [pixel][c]
    __shared__ int   msp[16 * KT][4];                      // corner spatial idx * C
    __shared__ float mw [16 * KT][4];                      // corner bilinear weights

    const int t    = threadIdx.x;
    const int P0   = blockIdx.x * 16;          // first pixel of tile
    const int b    = P0 >> 12;                 // / (Ho*Wo)
    const int rem  = P0 & 4095;
    const int ho   = rem >> 6;
    const int wo0  = rem & 63;

    const int lane = t & 31;
    const int wv   = t >> 5;
    const int col  = lane & 15;      // D column (pixel) / fragment row index
    const int h    = lane >> 4;      // lane half selects K sub-block
    const int n0a  = wv * 32;        // output-channel tiles for this wave
    const int n0b  = wv * 32 + 16;

    // scalar (SALU) predicate: exactly one wave issues TDM; no EXEC masking,
    // since TDM ignores EXEC and a masked-but-issued op would duplicate loads.
    const bool tdm_wave = (__builtin_amdgcn_readfirstlane(t) < 32);

#if HAVE_TDM
    if (tdm_wave)   // kick off tap-0 weight slab ASAP, overlaps metadata phase
        tdm_load_slab((unsigned)(uintptr_t)(const void*)&Bl[0][0], wb);
#endif

    // ---- phase 0: bilinear corner metadata for 16 pixels x 9 taps ----
    for (int e = t; e < 16 * KT; e += 128) {
        int m = e / KT, k = e - m * KT;
        int wo = wo0 + m;
        float dy = offs[(((size_t)b * 18 + 2 * k    ) * 64 + ho) * 64 + wo];
        float dx = offs[(((size_t)b * 18 + 2 * k + 1) * 64 + ho) * 64 + wo];
        float py = (float)(ho - 1 + (k / 3)) + dy;   // stride 1, pad 1, dil 1
        float px = (float)(wo - 1 + (k % 3)) + dx;
        float y0f = floorf(py), x0f = floorf(px);
        float ly = py - y0f, lx = px - x0f;
        int y0 = (int)y0f, x0 = (int)x0f;
        float wts[4] = { (1.f - ly) * (1.f - lx), (1.f - ly) * lx,
                         ly * (1.f - lx),          ly * lx };
        int ys[4] = { y0, y0, y0 + 1, y0 + 1 };
        int xs[4] = { x0, x0 + 1, x0, x0 + 1 };
        #pragma unroll
        for (int i = 0; i < 4; ++i) {
            bool valid = (ys[i] >= 0) && (ys[i] < Hh) && (xs[i] >= 0) && (xs[i] < Ww);
            int yc = min(max(ys[i], 0), Hh - 1);
            int xc = min(max(xs[i], 0), Ww - 1);
            msp[e][i] = (yc * Ww + xc) * Cch;          // pre-scaled by C
            mw [e][i] = valid ? wts[i] : 0.0f;
        }
    }

    v8f acc0 = {};   // persists over all taps / K-steps
    v8f acc1 = {};

    const unsigned short* xb = xt + (size_t)b * HWC;
    const int cp = (t & 63) * 2;     // channel pair for A-build
    const int mg = (t >> 6) * 8;     // pixel group (8 pixels) for A-build

    for (int k = 0; k < KT; ++k) {
        const int buf = k & 1;
        __syncthreads();   // previous iteration's LDS readers done (Al + both B bufs)

#if !HAVE_TDM
        // fallback: synchronous staging of weight slab
        {
            const uint4* src = (const uint4*)(wb + (size_t)k * SLAB);
            uint4* dst = (uint4*)&Bl[buf][0];
            #pragma unroll
            for (int r = 0; r < 16; ++r)
                dst[t + 128 * r] = src[t + 128 * r];
        }
#endif
        // ---- build sampled A slab: 2 channels x 8 pixels per thread, packed b32 ----
        #pragma unroll
        for (int mm = 0; mm < 8; ++mm) {
            const int m = mg + mm;
            const int e = m * KT + k;
            float a0 = 0.f, a1 = 0.f;
            #pragma unroll
            for (int i = 0; i < 4; ++i) {
                unsigned int u = *(const unsigned int*)(xb + msp[e][i] + cp);
                a0 += mw[e][i] * bflo2f(u);
                a1 += mw[e][i] * bfhi2f(u);
            }
            *(unsigned int*)&Al[m * Cch + cp] = pk_bf16(a0, a1);
        }

#if HAVE_TDM
        if (tdm_wave) {
            if (k + 1 < KT) {  // prefetch next slab into the other buffer (overlaps WMMA)
                tdm_load_slab((unsigned)(uintptr_t)(const void*)&Bl[buf ^ 1][0],
                              wb + (size_t)(k + 1) * SLAB);
                WAIT_TENSOR(1);   // slab k done (in-order), slab k+1 may stay in flight
            } else {
                WAIT_TENSOR(0);
            }
        }
#endif
        __syncthreads();

        // ---- 4 K-steps of v_wmma_f32_16x16x32_bf16 per wave ----
        #pragma unroll
        for (int s = 0; s < 4; ++s) {
            const int c0 = s * 32;
            // A fragments (weights, 16x32): row = o_local = col,
            // K = h*8 + j (j<8) and 16 + h*8 + (j-8)  -> two 16B LDS reads each
            v8bf a0lo = *(const v8bf*)&Bl[buf][(n0a + col) * Cch + c0 + h * 8];
            v8bf a0hi = *(const v8bf*)&Bl[buf][(n0a + col) * Cch + c0 + 16 + h * 8];
            v8bf a1lo = *(const v8bf*)&Bl[buf][(n0b + col) * Cch + c0 + h * 8];
            v8bf a1hi = *(const v8bf*)&Bl[buf][(n0b + col) * Cch + c0 + 16 + h * 8];
            v16bf a0 = __builtin_shufflevector(a0lo, a0hi,
                           0,1,2,3,4,5,6,7,8,9,10,11,12,13,14,15);
            v16bf a1 = __builtin_shufflevector(a1lo, a1hi,
                           0,1,2,3,4,5,6,7,8,9,10,11,12,13,14,15);
            // B fragment (sampled, 32x16): N = pixel = col, K = h*8 + j (j<8) and 16 + h*8 + (j-8) -> two 16B reads
            v8bf blo = *(const v8bf*)&Al[col * Cch + c0 + h * 8];
            v8bf bhi = *(const v8bf*)&Al[col * Cch + c0 + 16 + h * 8];
            v16bf bfrag = __builtin_shufflevector(blo, bhi,
                           0,1,2,3,4,5,6,7,8,9,10,11,12,13,14,15);
            acc0 = __builtin_amdgcn_wmma_f32_16x16x32_bf16(
                       false, a0, false, bfrag, (short)0, acc0, false, false);
            acc1 = __builtin_amdgcn_wmma_f32_16x16x32_bf16(
                       false, a1, false, bfrag, (short)0, acc1, false, false);
        }
    }

    // ---- epilogue: D rows = o, cols = pixel -> coalesced stores per VGPR ----
    const int wo = wo0 + col;
    float* outb = out + (size_t)b * Och * HWSZ + ho * 64 + wo;
    #pragma unroll
    for (int r = 0; r < 8; ++r) {
        const int oA = n0a + h * 8 + r;    // lanes 0-15: M=r, lanes 16-31: M=8+r
        const int oB = n0b + h * 8 + r;
        outb[(size_t)oA * HWSZ] = acc0[r] + bias[oA];
        outb[(size_t)oB * HWSZ] = acc1[r] + bias[oB];
    }
}

extern "C" void kernel_launch(void* const* d_in, const int* in_sizes, int n_in,
                              void* d_out, int out_size, void* d_ws, size_t ws_size,
                              hipStream_t stream) {
    const float* x    = (const float*)d_in[0];
    const float* offs = (const float*)d_in[1];
    const float* wgt  = (const float*)d_in[2];
    const float* bias = (const float*)d_in[3];
    float* out = (float*)d_out;

    // workspace: xt (B*H*W*C bf16 = 8.0 MiB) then Wb (K*O*C bf16 = 288 KiB)
    unsigned short* xt = (unsigned short*)d_ws;
    unsigned short* wb = xt + (size_t)Bsz * HWC;

    xpose_kernel<<<(Bsz * HWC) / 256, 256, 0, stream>>>(x, xt);
    wprep_kernel<<<(KT * Och * Cch + 255) / 256, 256, 0, stream>>>(wgt, wb);
    dcn_kernel<<<(Bsz * HWSZ) / 16, 128, 0, stream>>>(xt, offs, wb, bias, out);
}
